// CombinedLoss_42717744726103
// MI455X (gfx1250) — compile-verified, matrix operand order checked by reading, not run on hardware
//
#include <hip/hip_runtime.h>
#include <hip/hip_bf16.h>

typedef __attribute__((ext_vector_type(16))) _Float16 v16h;
typedef __attribute__((ext_vector_type(8)))  float    v8f;

#define EPSF 1e-7f
#define TWO_PI 6.28318530717958647692f

// K index inside a 16-bit 16x32 A/B operand for element e (0..15) of the v16h,
// per CDNA5 ISA 7.12.2: VGPR0..3 hold K=0..7 (lanes 0-15) / K=8..15 (lanes 16-31),
// VGPR4..7 hold K=16..23 / K=24..31.
__device__ __forceinline__ int kmap(int e, int lane) {
    return (e & 7) + ((e >> 3) << 4) + ((lane >> 4) << 3);
}

// ---------------------------------------------------------------------------
// zero the 16-float accumulator block
// ---------------------------------------------------------------------------
__global__ void init_acc(float* acc) {
    if (threadIdx.x < 16) acc[threadIdx.x] = 0.0f;
}

// ---------------------------------------------------------------------------
// Pre-swizzle DFT cos/sin matrices into WMMA B-operand register layout:
// storage index = ((ntile*ksteps + ks)*32 + lane)*16 + e
// value at (K = ks*32 + kmap(e,lane), N = ntile*16 + lane&15)
// Columns beyond n_bins are zero.
// ---------------------------------------------------------------------------
__global__ void prep_dft(_Float16* __restrict__ ctab, _Float16* __restrict__ stab,
                         int nfft, int nbins, int ntiles, int ksteps) {
    int idx = blockIdx.x * blockDim.x + threadIdx.x;
    int total = ntiles * ksteps * 512;
    if (idx >= total) return;
    int e    = idx & 15;
    int lane = (idx >> 4) & 31;
    int ks   = (idx >> 9) % ksteps;
    int nt   = idx / (ksteps * 512);
    int n = nt * 16 + (lane & 15);
    int k = ks * 32 + kmap(e, lane);
    float c = 0.0f, s = 0.0f;
    if (n < nbins) {
        int m = (k * n) % nfft;                 // exact phase reduction in ints
        float ph = TWO_PI * (float)m / (float)nfft;
        __sincosf(ph, &s, &c);
    }
    ctab[idx] = (_Float16)c;
    stab[idx] = (_Float16)s;
}

// ---------------------------------------------------------------------------
// Materialize windowed, reflect-padded frames in WMMA A-operand layout (f16).
// storage index = (((b*mtiles + mt)*ksteps + ks)*32 + lane)*16 + e
// value = x[b, reflect(frame*hop + s - pad)] * hann(s), frame = mt*16+(lane&15),
// s = ks*32 + kmap(e,lane). Rows past n_frames are zero (masked tail).
// Each sample is loaded & windowed exactly ONCE (vs. ntiles times if done in
// the GEMM loop).
// ---------------------------------------------------------------------------
__global__ void prep_frames(const float* __restrict__ pred,
                            const float* __restrict__ targ,
                            _Float16* __restrict__ fp, _Float16* __restrict__ ft,
                            int nfft, int hop, int nfr,
                            int mtiles, int ksteps, int S) {
    int idx = blockIdx.x * blockDim.x + threadIdx.x;
    int per_b = mtiles * ksteps * 512;
    int b = blockIdx.y;
    if (idx >= per_b) return;
    int e    = idx & 15;
    int lane = (idx >> 4) & 31;
    int ks   = (idx >> 9) % ksteps;
    int mt   = idx / (ksteps * 512);

    int frame = mt * 16 + (lane & 15);
    int s = ks * 32 + kmap(e, lane);
    float vp = 0.0f, vt = 0.0f;
    if (frame < nfr) {
        int pad = nfft >> 1;
        int pos = frame * hop + s - pad;          // reflect padding
        if (pos < 0)  pos = -pos;
        if (pos >= S) pos = 2 * S - 2 - pos;
        float w = 0.5f - 0.5f * __cosf(TWO_PI * (float)s / (float)nfft);
        size_t sig = (size_t)b * (size_t)S + (size_t)pos;
        vp = pred[sig] * w;
        vt = targ[sig] * w;
    }
    size_t o = (size_t)b * (size_t)per_b + (size_t)idx;
    fp[o] = (_Float16)vp;
    ft[o] = (_Float16)vt;
}

// ---------------------------------------------------------------------------
// Pure WMMA DFT-GEMM: one wave computes one 16x16 (frame x bin) output tile.
// Per K-step: 4 x 32B operand loads (A pred/targ + B cos/sin) and 4 WMMAs
// (B tiles shared across both signals) -> 16 flop/byte from L2, ~80 live
// VGPRs so the compiler can pipeline loads 2-deep with NO scratch spills.
// 8 waves per block. acc: [r]=sum((tm-pm)^2), [4+r]=sum(tm^2), [8+r]=sum|logr|
// ---------------------------------------------------------------------------
__global__ void stft_wmma(const _Float16* __restrict__ fp,
                          const _Float16* __restrict__ ft,
                          const _Float16* __restrict__ ctab,
                          const _Float16* __restrict__ stab,
                          float* acc, int r,
                          int nfr, int nbins,
                          int mtiles, int ntiles, int ksteps) {
    int lane = threadIdx.x & 31;
    int wave = threadIdx.x >> 5;
    int tile = blockIdx.x * 8 + wave;
    if (tile >= mtiles * ntiles) return;        // wave-uniform: EXEC stays all-1s

    int mt = tile / ntiles;
    int nt = tile % ntiles;
    int b  = blockIdx.y;

    const _Float16* aP = fp + (((size_t)b * mtiles + mt) * ksteps) * 512 + (size_t)lane * 16;
    const _Float16* aT = ft + (((size_t)b * mtiles + mt) * ksteps) * 512 + (size_t)lane * 16;
    const _Float16* bC = ctab + ((size_t)nt * ksteps) * 512 + (size_t)lane * 16;
    const _Float16* bS = stab + ((size_t)nt * ksteps) * 512 + (size_t)lane * 16;

    v8f pr = {}, pi = {}, tr = {}, ti = {};

    for (int ks = 0; ks < ksteps; ++ks) {
        size_t o = (size_t)ks * 512;
        v16h Ap = *(const v16h*)(aP + o);
        v16h At = *(const v16h*)(aT + o);
        v16h Bc = *(const v16h*)(bC + o);
        v16h Bs = *(const v16h*)(bS + o);
        if (ks + 1 < ksteps) {                  // global_prefetch_b8 next tiles
            __builtin_prefetch(aP + o + 512, 0, 0);
            __builtin_prefetch(bC + o + 512, 0, 0);
        }
        pr = __builtin_amdgcn_wmma_f32_16x16x32_f16(false, Ap, false, Bc, (short)0, pr, false, false);
        pi = __builtin_amdgcn_wmma_f32_16x16x32_f16(false, Ap, false, Bs, (short)0, pi, false, false);
        tr = __builtin_amdgcn_wmma_f32_16x16x32_f16(false, At, false, Bc, (short)0, tr, false, false);
        ti = __builtin_amdgcn_wmma_f32_16x16x32_f16(false, At, false, Bs, (short)0, ti, false, false);
    }

    // C/D layout: VGPR j, lane L -> M = mt*16 + j + 8*(L>=16), N = nt*16 + (L&15)
    float d2 = 0.0f, t2 = 0.0f, al = 0.0f;
    int N = nt * 16 + (lane & 15);
    int mofs = mt * 16 + ((lane >> 4) << 3);
    bool ncol = N < nbins;
    #pragma unroll
    for (int j = 0; j < 8; ++j) {
        int M = mofs + j;
        if (ncol && M < nfr) {
            float pm = sqrtf(pr[j] * pr[j] + pi[j] * pi[j]);
            float tm = sqrtf(tr[j] * tr[j] + ti[j] * ti[j]);
            float d = tm - pm;
            d2 += d * d;
            t2 += tm * tm;
            al += fabsf(__logf(pm + EPSF) - __logf(tm + EPSF));
        }
    }
    #pragma unroll
    for (int off = 16; off; off >>= 1) {
        d2 += __shfl_xor(d2, off, 32);
        t2 += __shfl_xor(t2, off, 32);
        al += __shfl_xor(al, off, 32);
    }
    if (lane == 0) {
        atomicAdd(acc + r,     d2);
        atomicAdd(acc + 4 + r, t2);
        atomicAdd(acc + 8 + r, al);
    }
}

// ---------------------------------------------------------------------------
// 8th-order IIR (direct form II transposed) over time; one lane per batch row,
// filters pred & target simultaneously, accumulates sum|pw - tw| -> acc[12]
// ---------------------------------------------------------------------------
__global__ void lfilter_kernel(const float* __restrict__ pred,
                               const float* __restrict__ targ,
                               const float* __restrict__ fb,
                               const float* __restrict__ fa,
                               float* acc, int B, int S) {
    int b = blockIdx.x * blockDim.x + threadIdx.x;
    if (b >= B) return;
    float bn[9], an[9];
    float inv_a0 = 1.0f / fa[0];
    #pragma unroll
    for (int i = 0; i < 9; ++i) { bn[i] = fb[i] * inv_a0; an[i] = fa[i] * inv_a0; }
    float zp[8] = {0}, zt[8] = {0};
    const float* xp = pred + (size_t)b * S;
    const float* xt = targ + (size_t)b * S;
    float sum = 0.0f;
    for (int t = 0; t < S; ++t) {
        float x = xp[t];
        float y = fmaf(bn[0], x, zp[0]);
        #pragma unroll
        for (int i = 0; i < 7; ++i) zp[i] = zp[i + 1] + bn[i + 1] * x - an[i + 1] * y;
        zp[7] = bn[8] * x - an[8] * y;

        float x2 = xt[t];
        float y2 = fmaf(bn[0], x2, zt[0]);
        #pragma unroll
        for (int i = 0; i < 7; ++i) zt[i] = zt[i + 1] + bn[i + 1] * x2 - an[i + 1] * y2;
        zt[7] = bn[8] * x2 - an[8] * y2;

        sum += fabsf(y - y2);
    }
    atomicAdd(acc + 12, sum);
}

// ---------------------------------------------------------------------------
// Loudness: per-batch RMS in dB, |diff| summed into acc[13]
// ---------------------------------------------------------------------------
__global__ void loud_kernel(const float* __restrict__ pred,
                            const float* __restrict__ targ,
                            float* acc, int S) {
    __shared__ float sp[256], st[256];
    size_t base = (size_t)blockIdx.x * (size_t)S;
    float s1 = 0.0f, s2 = 0.0f;
    for (int i = threadIdx.x; i < S; i += blockDim.x) {
        float v = pred[base + i]; s1 += v * v;
        float w = targ[base + i]; s2 += w * w;
    }
    sp[threadIdx.x] = s1; st[threadIdx.x] = s2;
    __syncthreads();
    for (int off = blockDim.x >> 1; off; off >>= 1) {
        if (threadIdx.x < off) {
            sp[threadIdx.x] += sp[threadIdx.x + off];
            st[threadIdx.x] += st[threadIdx.x + off];
        }
        __syncthreads();
    }
    if (threadIdx.x == 0) {
        float dbp = 20.0f * log10f(sqrtf(sp[0] / (float)S) + EPSF);
        float dbt = 20.0f * log10f(sqrtf(st[0] / (float)S) + EPSF);
        atomicAdd(acc + 13, fabsf(dbp - dbt));
    }
}

// ---------------------------------------------------------------------------
// Parameter penalties: sum|gains| -> acc[14], sum (q-1)^2 -> acc[15]
// ---------------------------------------------------------------------------
__global__ void param_kernel(const float* __restrict__ gains,
                             const float* __restrict__ q,
                             float* acc, int n) {
    __shared__ float sg[256], sq[256];
    float g = 0.0f, qq = 0.0f;
    for (int i = threadIdx.x; i < n; i += blockDim.x) {
        g += fabsf(gains[i]);
        float d = q[i] - 1.0f;
        qq += d * d;
    }
    sg[threadIdx.x] = g; sq[threadIdx.x] = qq;
    __syncthreads();
    for (int off = blockDim.x >> 1; off; off >>= 1) {
        if (threadIdx.x < off) {
            sg[threadIdx.x] += sg[threadIdx.x + off];
            sq[threadIdx.x] += sq[threadIdx.x + off];
        }
        __syncthreads();
    }
    if (threadIdx.x == 0) {
        atomicAdd(acc + 14, sg[0]);
        atomicAdd(acc + 15, sq[0]);
    }
}

// ---------------------------------------------------------------------------
// Combine everything into the scalar loss
// ---------------------------------------------------------------------------
__global__ void final_kernel(const float* __restrict__ acc, float* out,
                             int B, int S, int NB) {
    if (threadIdx.x != 0 || blockIdx.x != 0) return;
    const int nfr[4]  = {129, 257, 513, 1025};
    const int nbin[4] = {1025, 513, 257, 129};
    float l_spec = 0.0f;
    #pragma unroll
    for (int r = 0; r < 4; ++r) {
        float sc = sqrtf(acc[r]) / sqrtf(acc[4 + r]);
        float lg = acc[8 + r] / ((float)B * (float)nfr[r] * (float)nbin[r]);
        l_spec += sc + lg;
    }
    l_spec *= 0.25f;
    float l_perc  = acc[12] / ((float)B * (float)S);
    float l_loud  = acc[13] / (float)B;
    float l_param = 0.01f * acc[14] / (float)(B * NB) + 0.001f * acc[15] / (float)(B * NB);
    out[0] = 1.0f * l_spec + 0.1f * l_perc + 0.1f * l_loud + 1.0f * l_param;
}

// ---------------------------------------------------------------------------
extern "C" void kernel_launch(void* const* d_in, const int* in_sizes, int n_in,
                              void* d_out, int out_size, void* d_ws, size_t ws_size,
                              hipStream_t stream) {
    (void)in_sizes; (void)n_in; (void)out_size; (void)ws_size;
    const float* pred  = (const float*)d_in[0];
    const float* targ  = (const float*)d_in[1];
    const float* gains = (const float*)d_in[2];
    const float* qf    = (const float*)d_in[3];
    const float* fb    = (const float*)d_in[4];
    const float* fa    = (const float*)d_in[5];
    const int B = 32, S = 65536, NB = 10;
    const int NFFT[4] = {2048, 1024, 512, 256};
    const int HOP[4]  = {512, 256, 128, 64};

    float* acc = (float*)d_ws;                          // 16 f32 accumulators
    _Float16* tab = (_Float16*)((char*)d_ws + 256);     // DFT tables (~11.4 MB)

    init_acc<<<1, 32, 0, stream>>>(acc);

    _Float16* ct[4]; _Float16* st4[4];
    int nfr[4], nbin[4], mtl[4], ntl[4], kst[4];
    size_t off = 0;
    size_t maxFrame = 0;
    for (int r = 0; r < 4; ++r) {
        nfr[r]  = 1 + S / HOP[r];
        nbin[r] = NFFT[r] / 2 + 1;
        mtl[r]  = (nfr[r] + 15) / 16;
        ntl[r]  = (nbin[r] + 15) / 16;
        kst[r]  = NFFT[r] / 32;
        ct[r]  = tab + off; off += (size_t)ntl[r] * kst[r] * 512;
        st4[r] = tab + off; off += (size_t)ntl[r] * kst[r] * 512;
        size_t fr = (size_t)B * mtl[r] * kst[r] * 512;
        if (fr > maxFrame) maxFrame = fr;
        int total = ntl[r] * kst[r] * 512;
        prep_dft<<<(total + 255) / 256, 256, 0, stream>>>(ct[r], st4[r],
                                                          NFFT[r], nbin[r], ntl[r], kst[r]);
    }
    _Float16* fp = tab + off;                           // frame buffers, reused per r
    _Float16* ft = fp + maxFrame;                       // (~19 MB each at r=0)

    for (int r = 0; r < 4; ++r) {
        int per_b = mtl[r] * kst[r] * 512;
        dim3 pgrid((per_b + 255) / 256, B);
        prep_frames<<<pgrid, 256, 0, stream>>>(pred, targ, fp, ft,
                                               NFFT[r], HOP[r], nfr[r],
                                               mtl[r], kst[r], S);
        int tiles = mtl[r] * ntl[r];
        dim3 grid((tiles + 7) / 8, B);
        stft_wmma<<<grid, 256, 0, stream>>>(fp, ft, ct[r], st4[r], acc, r,
                                            nfr[r], nbin[r],
                                            mtl[r], ntl[r], kst[r]);
    }
    lfilter_kernel<<<1, 32, 0, stream>>>(pred, targ, fb, fa, acc, B, S);
    loud_kernel<<<B, 256, 0, stream>>>(pred, targ, acc, S);
    param_kernel<<<1, 256, 0, stream>>>(gains, qf, acc, B * NB);
    final_kernel<<<1, 1, 0, stream>>>(acc, (float*)d_out, B, S, NB);
}
